// DrW_30494267801752
// MI455X (gfx1250) — compile-verified, matrix operand order checked by reading, not run on hardware
//
#include <hip/hip_runtime.h>
#include <hip/hip_bf16.h>

// Problem sizes (match reference)
constexpr int Bn = 512, Ln = 10, Rn = 2048, Dn = 256, Kn = 20, Hn = 128;
constexpr int VOC = 30000;
constexpr int MROWS = 16;   // padded score rows (M dim of WMMA tile)

typedef _Float16 v16h __attribute__((ext_vector_type(16)));
typedef _Float16 v8h  __attribute__((ext_vector_type(8)));
typedef float    v8f  __attribute__((ext_vector_type(8)));

static __device__ __forceinline__ float wave_sum32(float v) {
#pragma unroll
  for (int o = 16; o > 0; o >>= 1) v += __shfl_xor(v, o, 32);
  return v;
}

// ---------------------------------------------------------------------------
// Kernel 1: build L2-normalized fp16 embedding table (one wave per vocab row)
// ---------------------------------------------------------------------------
__global__ void k_prep_norm_f16(const float* __restrict__ emb,
                                _Float16* __restrict__ embf16) {
  const int row  = blockIdx.x * 8 + (threadIdx.x >> 5);
  const int lane = threadIdx.x & 31;
  const float* r = emb + (size_t)row * Dn + lane * 8;
  const float4 a = ((const float4*)r)[0];
  const float4 b = ((const float4*)r)[1];
  float ss = a.x * a.x + a.y * a.y + a.z * a.z + a.w * a.w +
             b.x * b.x + b.y * b.y + b.z * b.z + b.w * b.w;
  ss = wave_sum32(ss);
  const float rn = 1.0f / fmaxf(sqrtf(ss), 1e-12f);
  v8h h;
  h[0] = (_Float16)(a.x * rn); h[1] = (_Float16)(a.y * rn);
  h[2] = (_Float16)(a.z * rn); h[3] = (_Float16)(a.w * rn);
  h[4] = (_Float16)(b.x * rn); h[5] = (_Float16)(b.y * rn);
  h[6] = (_Float16)(b.z * rn); h[7] = (_Float16)(b.w * rn);
  *(v8h*)(embf16 + (size_t)row * Dn + lane * 8) = h;
}

// ---------------------------------------------------------------------------
// Kernel 2: cosine-similarity GEMM via WMMA. One block (8 waves) per batch.
// Each wave owns a 256-column stripe of the 2048 docs (16 tiles of 16 cols).
// B fragments for the whole K=256 are preloaded as one load clause, then the
// 8-WMMA chain runs; D is stored unconditionally to a 16-row padded buffer.
// ---------------------------------------------------------------------------
__global__ void k_match_wmma(const int* __restrict__ text_left,
                             const int* __restrict__ text_right,
                             const _Float16* __restrict__ embf16,
                             float* __restrict__ matching) {
  constexpr int APAD = Dn + 8;                 // 8-f16 (=16B-aligned) row pad
  __shared__ _Float16 As[16][APAD];

  const int b    = blockIdx.x;
  const int tid  = threadIdx.x;
  const int wv   = tid >> 5;
  const int lane = tid & 31;

  // Stage normalized query tile (rows 10..15 zeroed)
  {
    const int row = tid >> 4, chunk = tid & 15;
    if (row < Ln) {
      const int tok = text_left[b * Ln + row];
      const _Float16* src = embf16 + (size_t)tok * Dn + chunk * 16;
#pragma unroll
      for (int i = 0; i < 16; ++i) As[row][chunk * 16 + i] = src[i];
    } else {
#pragma unroll
      for (int i = 0; i < 16; ++i) As[row][chunk * 16 + i] = (_Float16)0.0f;
    }
  }
  __syncthreads();

  const int n     = lane & 15;     // tile column (B matrix N)  / A matrix row M
  const int khalf = lane >> 4;     // K-half selector per ISA 16-bit layout

  // Hoist A fragments (shared by all 16 tiles of this wave)
  v16h af[8];
#pragma unroll
  for (int kk = 0; kk < 8; ++kk) {
    const v8h alo = *(const v8h*)&As[n][kk * 32 + khalf * 8];
    const v8h ahi = *(const v8h*)&As[n][kk * 32 + 16 + khalf * 8];
    af[kk] = __builtin_shufflevector(
        alo, ahi, 0, 1, 2, 3, 4, 5, 6, 7, 8, 9, 10, 11, 12, 13, 14, 15);
  }

  for (int t = 0; t < 16; ++t) {
    const int col = wv * 256 + t * 16 + n;
    const int tok = text_right[b * Rn + col];
    const _Float16* brow = embf16 + (size_t)tok * Dn + khalf * 8;

    // Preload the full 512B doc row as 16 x b128 (one clause, overlapped)
    v8h blo[8], bhi[8];
#pragma unroll
    for (int kk = 0; kk < 8; ++kk) {
      blo[kk] = *(const v8h*)(brow + kk * 32);
      bhi[kk] = *(const v8h*)(brow + kk * 32 + 16);
    }

    v8f acc = {};
#pragma unroll
    for (int kk = 0; kk < 8; ++kk) {          // K = 256 in steps of 32
      const v16h bfrag = __builtin_shufflevector(
          blo[kk], bhi[kk],
          0, 1, 2, 3, 4, 5, 6, 7, 8, 9, 10, 11, 12, 13, 14, 15);
      acc = __builtin_amdgcn_wmma_f32_16x16x32_f16(
          false, af[kk], false, bfrag, (short)0, acc, false, false);
    }

    // D layout: VGPR v -> row M = v + 8*khalf, col N = lane&15.
    // Store all 16 rows (10 real + 6 pad) -> branch-free, uniform stores.
    float* drow = matching + ((size_t)b * MROWS + 8 * khalf) * Rn
                           + wv * 256 + t * 16 + n;
#pragma unroll
    for (int v = 0; v < 8; ++v) drow[(size_t)v * Rn] = acc[v];
  }
}

// ---------------------------------------------------------------------------
// Kernel 3: top-20 + MLP + attention + geo head. 10 waves per batch.
// ---------------------------------------------------------------------------
__global__ void k_finalize(const int* __restrict__ text_left,
                           const int* __restrict__ loc,
                           const float* __restrict__ distance,
                           const float* __restrict__ emb,
                           const float* __restrict__ attn_w,
                           const float* __restrict__ w_m0, const float* __restrict__ b_m0,
                           const float* __restrict__ w_m1, const float* __restrict__ b_m1,
                           const float* __restrict__ w_m2, const float* __restrict__ b_m2,
                           const float* __restrict__ out_w, const float* __restrict__ out_b,
                           const float* __restrict__ lat_tab, const float* __restrict__ lon_tab,
                           const float* __restrict__ w3, const float* __restrict__ b3,
                           const float* __restrict__ w5, const float* __restrict__ b5,
                           const float* __restrict__ w4, const float* __restrict__ b4,
                           const float* __restrict__ matching,
                           float* __restrict__ pred) {
  __shared__ float sh_h0[Ln][Hn];
  __shared__ float sh_h1[Ln][Hn];
  __shared__ float sh_h2[Ln];
  __shared__ float sh_logit[Ln];
  __shared__ float sh_q1[Ln * 10];
  __shared__ float sh_lat[32];
  __shared__ float sh_lon[32];

  const int b    = blockIdx.x;
  const int l    = threadIdx.x >> 5;   // wave id == query position, 0..9
  const int lane = threadIdx.x & 31;
  const float NEG_INF = -__builtin_inff();

  // ---- per-lane sorted top-20 over 64 strided elements ----
  float t[Kn];
#pragma unroll
  for (int k = 0; k < Kn; ++k) t[k] = NEG_INF;
  const float* rowp = matching + ((size_t)b * MROWS + l) * Rn;
  for (int j = 0; j < Rn / 32; ++j) {
    const float val = rowp[lane + 32 * j];
#pragma unroll
    for (int k = Kn - 1; k >= 1; --k)
      t[k] = (val > t[k]) ? ((val > t[k - 1]) ? t[k - 1] : val) : t[k];
    t[0] = fmaxf(t[0], val);
  }
  // ---- 20-round cross-lane merge (descending, like lax.top_k) ----
  float tk[Kn];
#pragma unroll
  for (int r = 0; r < Kn; ++r) {
    float m = t[0];
    int mi = lane;
#pragma unroll
    for (int o = 16; o > 0; o >>= 1) {
      const float om = __shfl_xor(m, o, 32);
      const int omi  = __shfl_xor(mi, o, 32);
      if (om > m || (om == m && omi < mi)) { m = om; mi = omi; }
    }
    tk[r] = m;
    if (lane == mi) {
#pragma unroll
      for (int k = 0; k < Kn - 1; ++k) t[k] = t[k + 1];
      t[Kn - 1] = NEG_INF;
    }
  }

  // ---- MLP layer 0: 20 -> 128, tanh ----
#pragma unroll
  for (int i = 0; i < 4; ++i) {
    const int j = lane + 32 * i;
    float s = b_m0[j];
#pragma unroll
    for (int k = 0; k < Kn; ++k) s += tk[k] * w_m0[k * Hn + j];
    sh_h0[l][j] = tanhf(s);
  }
  __syncthreads();

  // ---- MLP layer 1: 128 -> 128, tanh ----
  float h1v[4];
#pragma unroll
  for (int i = 0; i < 4; ++i) {
    const int j = lane + 32 * i;
    float s = b_m1[j];
    for (int k = 0; k < Hn; ++k) s += sh_h0[l][k] * w_m1[k * Hn + j];
    h1v[i] = tanhf(s);
  }
  __syncthreads();
#pragma unroll
  for (int i = 0; i < 4; ++i) sh_h1[l][lane + 32 * i] = h1v[i];
  __syncthreads();

  // ---- MLP layer 2: 128 -> 1, tanh ----
  {
    float s = 0.f;
#pragma unroll
    for (int i = 0; i < 4; ++i) {
      const int j = lane + 32 * i;
      s += sh_h1[l][j] * w_m2[j];
    }
    s = wave_sum32(s);
    if (lane == 0) sh_h2[l] = tanhf(s + b_m2[0]);
  }

  // ---- attention logit + q1 slice (uses raw f32 embeddings) ----
  const int tok = text_left[b * Ln + l];
  const float* qrow = emb + (size_t)tok * Dn;
  {
    float s = 0.f;
#pragma unroll
    for (int i = 0; i < 8; ++i) {
      const int k = lane + 32 * i;
      s += qrow[k] * attn_w[k];
    }
    s = wave_sum32(s);
    if (lane == 0) sh_logit[l] = (tok == 0) ? NEG_INF : s;
  }
  if (lane < 10) {
    float s = b3[lane];
    for (int k = 0; k < Dn; ++k) s += qrow[k] * w3[k * 10 + lane];
    sh_q1[l * 10 + lane] = s;
  }

  // ---- geo embeddings (waves 0 and 1) ----
  if (l == 0) {
    const int g = loc[b * 2 + 0];
    float s = b5[lane];
    for (int k = 0; k < 64; ++k) s += lat_tab[g * 64 + k] * w5[k * 32 + lane];
    sh_lat[lane] = s;
  }
  if (l == 1) {
    const int g = loc[b * 2 + 1];
    float s = b5[lane];
    for (int k = 0; k < 64; ++k) s += lon_tab[g * 64 + k] * w5[k * 32 + lane];
    sh_lon[lane] = s;
  }
  __syncthreads();

  // ---- softmax, x, 164->2 head, final prediction ----
  if (threadIdx.x == 0) {
    float mx = NEG_INF;
    for (int i = 0; i < Ln; ++i) mx = fmaxf(mx, sh_logit[i]);
    float den = 0.f, num = 0.f;
    for (int i = 0; i < Ln; ++i) {
      const float e = expf(sh_logit[i] - mx);
      den += e;
      num += sh_h2[i] * e;
    }
    float x = num / den;
    x = tanhf(x * out_w[0] + out_b[0]);

    float w0 = b4[0], w1 = b4[1];
    for (int i = 0; i < 100; ++i) { w0 += sh_q1[i] * w4[i * 2]; w1 += sh_q1[i] * w4[i * 2 + 1]; }
    for (int j = 0; j < 32; ++j)  { w0 += sh_lat[j] * w4[(100 + j) * 2]; w1 += sh_lat[j] * w4[(100 + j) * 2 + 1]; }
    for (int j = 0; j < 32; ++j)  { w0 += sh_lon[j] * w4[(132 + j) * 2]; w1 += sh_lon[j] * w4[(132 + j) * 2 + 1]; }

    pred[b] = w0 * x + w1 * distance[b];
  }
}

// ---------------------------------------------------------------------------
extern "C" void kernel_launch(void* const* d_in, const int* in_sizes, int n_in,
                              void* d_out, int out_size, void* d_ws, size_t ws_size,
                              hipStream_t stream) {
  const int*   text_left  = (const int*)d_in[0];
  const int*   text_right = (const int*)d_in[1];
  const int*   loc        = (const int*)d_in[2];
  const float* distance   = (const float*)d_in[3];
  const float* emb        = (const float*)d_in[4];
  const float* attn_w     = (const float*)d_in[5];
  const float* w_m0       = (const float*)d_in[6];
  const float* b_m0       = (const float*)d_in[7];
  const float* w_m1       = (const float*)d_in[8];
  const float* b_m1       = (const float*)d_in[9];
  const float* w_m2       = (const float*)d_in[10];
  const float* b_m2       = (const float*)d_in[11];
  const float* out_w      = (const float*)d_in[12];
  const float* out_b      = (const float*)d_in[13];
  const float* lat_tab    = (const float*)d_in[14];
  const float* lon_tab    = (const float*)d_in[15];
  const float* w3         = (const float*)d_in[16];
  const float* b3         = (const float*)d_in[17];
  const float* w5         = (const float*)d_in[18];
  const float* b5         = (const float*)d_in[19];
  const float* w4         = (const float*)d_in[20];
  const float* b4         = (const float*)d_in[21];

  // Workspace layout: [normalized f16 table | padded matching scores]
  char* base = (char*)d_ws;
  _Float16* embf16 = (_Float16*)base;                       // 30000*256*2 B
  const size_t EMBF16_BYTES = (size_t)VOC * Dn * sizeof(_Float16);
  float* matching = (float*)(base + EMBF16_BYTES);          // 512*16*2048*4 B

  k_prep_norm_f16<<<dim3(VOC / 8), dim3(256), 0, stream>>>(emb, embf16);
  k_match_wmma<<<dim3(Bn), dim3(256), 0, stream>>>(text_left, text_right,
                                                   embf16, matching);
  k_finalize<<<dim3(Bn), dim3(Ln * 32), 0, stream>>>(
      text_left, loc, distance, emb, attn_w,
      w_m0, b_m0, w_m1, b_m1, w_m2, b_m2, out_w, out_b,
      lat_tab, lon_tab, w3, b3, w5, b5, w4, b4,
      matching, (float*)d_out);
}
